// PerceiverAttentionCA_54958401519757
// MI455X (gfx1250) — matrix operand order repaired and checked by reading, not compiled, not run on hardware
//
#include <hip/hip_runtime.h>
#include <hip/hip_bf16.h>

typedef __bf16 bf16;
typedef __attribute__((ext_vector_type(16))) __bf16 v16bf;
typedef __attribute__((ext_vector_type(8)))  __bf16 v8bf;
typedef __attribute__((ext_vector_type(8)))  float   v8f;

__device__ __forceinline__ unsigned short f2bfraw(float f) {
    union { float f; unsigned u; } v; v.f = f;
    unsigned r = v.u + 0x7fffu + ((v.u >> 16) & 1u);   // round-to-nearest-even
    return (unsigned short)(r >> 16);
}
__device__ __forceinline__ bf16 f2bf(float f) {
    union { unsigned short s; bf16 b; } u; u.s = f2bfraw(f);
    return u.b;
}

// ---------------------------------------------------------------------------
// LayerNorm over rows of length 3072, fp32 -> bf16.  One 256-thread block/row.
// ---------------------------------------------------------------------------
__global__ __launch_bounds__(256) void ln_bf16_kernel(
    const float* __restrict__ x, const float* __restrict__ gamma,
    const float* __restrict__ beta, bf16* __restrict__ y)
{
    const int row = blockIdx.x;
    const int tid = threadIdx.x;
    const float* xr = x + (size_t)row * 3072;

    float v[12];
    float s = 0.f, s2 = 0.f;
#pragma unroll
    for (int i = 0; i < 3; ++i) {
        float4 f = *(const float4*)(xr + i * 1024 + tid * 4);
        v[i*4+0]=f.x; v[i*4+1]=f.y; v[i*4+2]=f.z; v[i*4+3]=f.w;
        s  += f.x + f.y + f.z + f.w;
        s2 += f.x*f.x + f.y*f.y + f.z*f.z + f.w*f.w;
    }
    __shared__ float r1[256], r2[256];
    r1[tid] = s; r2[tid] = s2; __syncthreads();
    for (int w = 128; w > 0; w >>= 1) {
        if (tid < w) { r1[tid] += r1[tid+w]; r2[tid] += r2[tid+w]; }
        __syncthreads();
    }
    const float mean = r1[0] * (1.0f / 3072.0f);
    const float var  = r2[0] * (1.0f / 3072.0f) - mean * mean;
    const float rstd = rsqrtf(var + 1e-5f);

    bf16* yr = y + (size_t)row * 3072;
#pragma unroll
    for (int i = 0; i < 3; ++i) {
        int c = i * 1024 + tid * 4;
        unsigned h0 = f2bfraw((v[i*4+0]-mean)*rstd*gamma[c+0] + beta[c+0]);
        unsigned h1 = f2bfraw((v[i*4+1]-mean)*rstd*gamma[c+1] + beta[c+1]);
        unsigned h2 = f2bfraw((v[i*4+2]-mean)*rstd*gamma[c+2] + beta[c+2]);
        unsigned h3 = f2bfraw((v[i*4+3]-mean)*rstd*gamma[c+3] + beta[c+3]);
        uint2 pk; pk.x = h0 | (h1 << 16); pk.y = h2 | (h3 << 16);
        *(uint2*)(yr + c) = pk;
    }
}

// ---------------------------------------------------------------------------
// Cast + transpose:  dst[o*dstLd + c] = src[c*srcLd + colOff + o] * scale
// ---------------------------------------------------------------------------
__global__ __launch_bounds__(256) void cast_transpose_kernel(
    const float* __restrict__ src, int srcLd, int colOff,
    bf16* __restrict__ dst, int dstLd, float scale)
{
    __shared__ float tile[32][33];
    const int o0 = blockIdx.x * 32;
    const int c0 = blockIdx.y * 32;
    const int tx = threadIdx.x, ty = threadIdx.y;
#pragma unroll
    for (int j = 0; j < 4; ++j)
        tile[ty + 8*j][tx] = src[(size_t)(c0 + ty + 8*j) * srcLd + colOff + o0 + tx];
    __syncthreads();
#pragma unroll
    for (int j = 0; j < 4; ++j)
        dst[(size_t)(o0 + ty + 8*j) * dstLd + c0 + tx] = f2bf(tile[tx][ty + 8*j] * scale);
}

// ---------------------------------------------------------------------------
// Row softmax, rows of 1024 fp32 -> bf16.  One 256-thread block per row.
// ---------------------------------------------------------------------------
__global__ __launch_bounds__(256) void softmax_row_kernel(
    const float* __restrict__ S, bf16* __restrict__ P)
{
    const int row = blockIdx.x;
    const int tid = threadIdx.x;
    const float4 f = *(const float4*)(S + (size_t)row * 1024 + tid * 4);

    __shared__ float red[256];
    float m = fmaxf(fmaxf(f.x, f.y), fmaxf(f.z, f.w));
    red[tid] = m; __syncthreads();
    for (int w = 128; w > 0; w >>= 1) {
        if (tid < w) red[tid] = fmaxf(red[tid], red[tid + w]);
        __syncthreads();
    }
    m = red[0]; __syncthreads();

    const float e0 = __expf(f.x - m), e1 = __expf(f.y - m);
    const float e2 = __expf(f.z - m), e3 = __expf(f.w - m);
    red[tid] = e0 + e1 + e2 + e3; __syncthreads();
    for (int w = 128; w > 0; w >>= 1) {
        if (tid < w) red[tid] += red[tid + w];
        __syncthreads();
    }
    const float inv = 1.0f / red[0];

    uint2 pk;
    pk.x = (unsigned)f2bfraw(e0 * inv) | ((unsigned)f2bfraw(e1 * inv) << 16);
    pk.y = (unsigned)f2bfraw(e2 * inv) | ((unsigned)f2bfraw(e3 * inv) << 16);
    *(uint2*)(P + (size_t)row * 1024 + tid * 4) = pk;
}

// ---------------------------------------------------------------------------
// C(M,N) = A(M,K) @ Bt(N,K)^T   — bf16 in, f32 accumulate, WMMA bf16.
// Block tile 128x128x64, 8 waves (2x4), wave tile 64x32 (4x2 WMMA tiles).
// Double-buffered LDS filled with global_load_async_to_lds_b128 (ASYNCcnt),
// so global latency hides behind the previous tile's WMMAs.
// ---------------------------------------------------------------------------
__device__ __forceinline__ void storeC(float* p, float v) { *p = v; }
__device__ __forceinline__ void storeC(bf16*  p, float v) { *p = f2bf(v); }

template <typename OutT>
__global__ __launch_bounds__(256) void gemm_bt_kernel(
    const bf16* __restrict__ A, const bf16* __restrict__ B, OutT* __restrict__ C,
    int M, int N, int K,
    long long aStride, long long bStride, long long cStride)
{
    constexpr int LDT = 72;                   // padded LDS row stride (bf16)
    // [0]=A buf0, [1]=A buf1, [2]=B buf0, [3]=B buf1  (4 x 18KB = 72KB)
    __shared__ bf16 sm[4][128 * LDT];

    A += (size_t)blockIdx.z * aStride;
    B += (size_t)blockIdx.z * bStride;
    C += (size_t)blockIdx.z * cStride;

    const int tid   = threadIdx.x;
    const int wid   = tid >> 5;
    const int lane  = tid & 31;
    const int lhalf = lane >> 4;              // 0 | 1
    const int l16   = lane & 15;
    const int waveM = wid >> 2;               // 0..1  -> 64 rows each
    const int waveN = wid & 3;                // 0..3  -> 32 cols each

    const int mBase = blockIdx.y * 128;
    const int nBase = blockIdx.x * 128;

    // Per-thread staging coords: 128x64 bf16 tile = 1024 16B chunks, 4/thread.
    const bf16* pa[4];
    const bf16* pb[4];
    unsigned ldsOffA[4], ldsOffB[4];
#pragma unroll
    for (int r = 0; r < 4; ++r) {
        const int chunk = tid + 256 * r;
        const int row = chunk >> 3;
        const int col = (chunk & 7) * 8;
        pa[r] = A + (size_t)(mBase + row) * K + col;
        pb[r] = B + (size_t)(nBase + row) * K + col;
        ldsOffA[r] = (unsigned)(uintptr_t)&sm[0][row * LDT + col];
        ldsOffB[r] = (unsigned)(uintptr_t)&sm[2][row * LDT + col];
    }
    constexpr unsigned BUFB = 128 * LDT * 2;  // bytes per buffer

    v8f acc[4][2];
#pragma unroll
    for (int i = 0; i < 4; ++i)
#pragma unroll
        for (int j = 0; j < 2; ++j)
#pragma unroll
            for (int e = 0; e < 8; ++e) acc[i][j][e] = 0.0f;

    // --- async tile fetch: 8 x b128 per thread directly into LDS -----------
    auto issue = [&](int buf) {
#pragma unroll
        for (int r = 0; r < 4; ++r) {
            unsigned la = ldsOffA[r] + (unsigned)buf * BUFB;
            unsigned lb = ldsOffB[r] + (unsigned)buf * BUFB;
            asm volatile("global_load_async_to_lds_b128 %0, %1, off"
                         :: "v"(la), "v"(pa[r]) : "memory");
            asm volatile("global_load_async_to_lds_b128 %0, %1, off"
                         :: "v"(lb), "v"(pb[r]) : "memory");
            pa[r] += 64; pb[r] += 64;
        }
    };

    issue(0);
    int buf = 0;
    for (int k0 = 0; k0 < K; k0 += 64) {
        if (k0 + 64 < K) {
            issue(buf ^ 1);                           // prefetch next tile
            asm volatile("s_wait_asynccnt 0x8" ::: "memory"); // current tile done
        } else {
            asm volatile("s_wait_asynccnt 0x0" ::: "memory");
        }
        __syncthreads();                              // tile visible to all waves

        const bf16* baseA = sm[buf];
        const bf16* baseB = sm[2 + buf];
#pragma unroll
        for (int kk = 0; kk < 64; kk += 32) {
            // A fragment: lane<16 -> K[0..7]+K[16..23]; lane>=16 -> K[8..15]+K[24..31]
            v16bf af[4];
#pragma unroll
            for (int mt = 0; mt < 4; ++mt) {
                const bf16* p = &baseA[(waveM * 64 + mt * 16 + l16) * LDT + kk + lhalf * 8];
                v8bf lo = *(const v8bf*)(const void*)p;
                v8bf hi = *(const v8bf*)(const void*)(p + 16);
                v16bf a;
#pragma unroll
                for (int e = 0; e < 8; ++e) { a[e] = lo[e]; a[8 + e] = hi[e]; }
                af[mt] = a;
            }
            // B fragment: lane<16 -> K[0..7]+K[16..23]; lane>=16 -> K[8..15]+K[24..31] (col = l16)
            v16bf bfr[2];
#pragma unroll
            for (int nt = 0; nt < 2; ++nt) {
                const bf16* p = &baseB[(waveN * 32 + nt * 16 + l16) * LDT + kk + lhalf * 8];
                v8bf lo = *(const v8bf*)(const void*)p;
                v8bf hi = *(const v8bf*)(const void*)(p + 16);
                v16bf b;
#pragma unroll
                for (int e = 0; e < 8; ++e) { b[e] = lo[e]; b[8 + e] = hi[e]; }
                bfr[nt] = b;
            }
#pragma unroll
            for (int mt = 0; mt < 4; ++mt)
#pragma unroll
                for (int nt = 0; nt < 2; ++nt)
                    acc[mt][nt] = __builtin_amdgcn_wmma_f32_16x16x32_bf16(
                        false, af[mt], false, bfr[nt], (short)0, acc[mt][nt],
                        false, false);
        }
        __syncthreads();                              // reads done; buffer reusable
        buf ^= 1;
    }

    // Writeback: c[v] lives at (row0 + v, l16) with row0 including 8*lhalf.
#pragma unroll
    for (int mt = 0; mt < 4; ++mt)
#pragma unroll
        for (int nt = 0; nt < 2; ++nt) {
            const int row0 = mBase + waveM * 64 + mt * 16 + lhalf * 8;
            const int col  = nBase + waveN * 32 + nt * 16 + l16;
#pragma unroll
            for (int v = 0; v < 8; ++v)
                storeC(&C[(size_t)(row0 + v) * N + col], acc[mt][nt][v]);
        }
}

// ---------------------------------------------------------------------------
extern "C" void kernel_launch(void* const* d_in, const int* in_sizes, int n_in,
                              void* d_out, int out_size, void* d_ws, size_t ws_size,
                              hipStream_t stream)
{
    const float* x   = (const float*)d_in[0];
    const float* lat = (const float*)d_in[1];
    const float* g1  = (const float*)d_in[2];
    const float* b1  = (const float*)d_in[3];
    const float* g2  = (const float*)d_in[4];
    const float* b2  = (const float*)d_in[5];
    const float* Wq  = (const float*)d_in[6];
    const float* Wkv = (const float*)d_in[7];
    const float* Wo  = (const float*)d_in[8];
    float* out = (float*)d_out;

    constexpr int T = 25, AA = 1024, HW = 2048, D = 3072, DH = 1024;
    const float qscale = 0.17677669529663687f;   // 1024^-0.25, folded into Wq & Wk

    size_t off = 0;
    auto alloc = [&](size_t elems) -> bf16* {
        bf16* p = (bf16*)((char*)d_ws + off);
        off += (elems * 2 + 255) & ~(size_t)255;
        return p;
    };
    bf16* WqT = alloc((size_t)DH * D);           // (DH, D)  with qscale folded
    bf16* WkT = alloc((size_t)DH * D);           // (DH, D)  with qscale folded
    bf16* WvT = alloc((size_t)DH * D);           // (DH, D)
    bf16* WoT = alloc((size_t)D * DH);           // (D, DH)
    bf16* xln = alloc((size_t)T * AA * D);       // LN(x)       bf16
    bf16* lln = alloc((size_t)T * HW * D);       // LN(latents) bf16
    bf16* Q   = alloc((size_t)T * HW * DH);
    bf16* Kb  = alloc((size_t)T * AA * DH);      // K rows (keys, d) — "Bt" form
    bf16* VT  = alloc((size_t)T * DH * AA);      // V^T (d, keys)   — "Bt" form
    // Aliased stage buffers (pipeline order makes these safe):
    float* S  = (float*)lln;                     // lln dead after Q-GEMM
    bf16*  P  = xln;                             // xln dead after K/VT GEMMs
    bf16*  O  = Q;                               // Q dead after S-GEMM

    const dim3 tb(32, 8);
    cast_transpose_kernel<<<dim3(DH/32, D/32), tb, 0, stream>>>(Wq,  DH,     0,  WqT, D,  qscale);
    cast_transpose_kernel<<<dim3(DH/32, D/32), tb, 0, stream>>>(Wkv, 2*DH,   0,  WkT, D,  qscale);
    cast_transpose_kernel<<<dim3(DH/32, D/32), tb, 0, stream>>>(Wkv, 2*DH,  DH,  WvT, D,  1.0f);
    cast_transpose_kernel<<<dim3(D/32, DH/32), tb, 0, stream>>>(Wo,  D,      0,  WoT, DH, 1.0f);

    ln_bf16_kernel<<<T * AA, 256, 0, stream>>>(x,   g1, b1, xln);
    ln_bf16_kernel<<<T * HW, 256, 0, stream>>>(lat, g2, b2, lln);

    // Q = lln @ Wq        (M=HW, N=DH, K=D)
    gemm_bt_kernel<bf16><<<dim3(DH/128, HW/128, T), 256, 0, stream>>>(
        lln, WqT, Q, HW, DH, D, (long long)HW * D, 0, (long long)HW * DH);
    // K = xln @ Wk        (M=AA, N=DH, K=D)
    gemm_bt_kernel<bf16><<<dim3(DH/128, AA/128, T), 256, 0, stream>>>(
        xln, WkT, Kb, AA, DH, D, (long long)AA * D, 0, (long long)AA * DH);
    // V^T = WvT @ xln^T   (M=DH, N=AA, K=D)
    gemm_bt_kernel<bf16><<<dim3(AA/128, DH/128, T), 256, 0, stream>>>(
        WvT, xln, VT, DH, AA, D, 0, (long long)AA * D, (long long)DH * AA);
    // S = Q @ K^T (fp32)  (M=HW, N=AA, K=DH)
    gemm_bt_kernel<float><<<dim3(AA/128, HW/128, T), 256, 0, stream>>>(
        Q, Kb, S, HW, AA, DH, (long long)HW * DH, (long long)AA * DH, (long long)HW * AA);
    // P = softmax(S)
    softmax_row_kernel<<<T * HW, 256, 0, stream>>>(S, P);
    // O = P @ V           (M=HW, N=DH, K=AA)
    gemm_bt_kernel<bf16><<<dim3(DH/128, HW/128, T), 256, 0, stream>>>(
        P, VT, O, HW, DH, AA, (long long)HW * AA, (long long)DH * AA, (long long)HW * DH);
    // out = O @ Wo (fp32) (M=HW, N=D, K=DH)
    gemm_bt_kernel<float><<<dim3(D/128, HW/128, T), 256, 0, stream>>>(
        O, WoT, out, HW, D, DH, (long long)HW * DH, 0, (long long)HW * D);
}
